// IntActWeight_6004364280068
// MI455X (gfx1250) — 8-bit matrix operand order repaired and checked by reading, compiled, not run on hardware
//
#include <hip/hip_runtime.h>


typedef int v4i __attribute__((ext_vector_type(4)));
typedef int v8i __attribute__((ext_vector_type(8)));

#define B_   8
#define M_   2048
#define K_   4096
#define N_   4096
#define KT64 (K_ / 64)
#define MT16 (M_ / 16)
#define NT16 (N_ / 16)

#define A_PACK_BYTES ((size_t)B_ * MT16 * KT64 * 1024)
#define B_PACK_BYTES ((size_t)NT16 * KT64 * 1024)

union AccU { v8i v; int s[8]; };

__global__ __launch_bounds__(256) void pack_a_kernel(const int* __restrict__ x,
                                                     unsigned char* __restrict__ ap) {
    long long t    = (long long)blockIdx.x * 256 + threadIdx.x;
    int       lane = (int)(t & 31);
    long long tile = t >> 5;
    int       kt   = (int)(tile % KT64);
    long long bm   = tile / KT64;
    int       mt   = (int)(bm % MT16);
    int       b    = (int)(bm / MT16);
    if (b >= B_) return;

    int m  = (mt << 4) + (lane & 15);
    int hi = lane >> 4;
    const v4i* src =
        (const v4i*)(x + ((long long)b * M_ + m) * K_ + ((long long)kt << 6));

    unsigned int packed[8];
#pragma unroll
    for (int g = 0; g < 8; ++g) {
        int kb4 = (((g >> 1) << 4) + (hi << 3) + ((g & 1) << 2)) >> 2;
        v4i d   = __builtin_nontemporal_load(src + kb4);
        packed[g] = ((unsigned)(d.x & 255)) |
                    ((unsigned)(d.y & 255) << 8) |
                    ((unsigned)(d.z & 255) << 16) |
                    ((unsigned)(d.w & 255) << 24);
    }
    v4i* dst = (v4i*)(ap + tile * 1024 + (long long)lane * 32);
    dst[0] = (v4i){(int)packed[0], (int)packed[1], (int)packed[2], (int)packed[3]};
    dst[1] = (v4i){(int)packed[4], (int)packed[5], (int)packed[6], (int)packed[7]};
}

__global__ __launch_bounds__(256) void pack_b_kernel(const int* __restrict__ y,
                                                     unsigned char* __restrict__ bp) {
    long long t    = (long long)blockIdx.x * 256 + threadIdx.x;
    int       lane = (int)(t & 31);
    long long tile = t >> 5;
    int       kt   = (int)(tile % KT64);
    int       nt   = (int)(tile / KT64);
    if (nt >= NT16) return;

    int n     = (nt << 4) + (lane & 15);
    int khalf = lane >> 4;
    const int* src = y + ((long long)kt << 6) * N_ + n;

    unsigned int packed[8];
#pragma unroll
    for (int g = 0; g < 8; ++g) {
        int kb = ((g >> 1) << 4) + (khalf << 3) + ((g & 1) << 2);
        int e0 = __builtin_nontemporal_load(src + (long long)(kb + 0) * N_);
        int e1 = __builtin_nontemporal_load(src + (long long)(kb + 1) * N_);
        int e2 = __builtin_nontemporal_load(src + (long long)(kb + 2) * N_);
        int e3 = __builtin_nontemporal_load(src + (long long)(kb + 3) * N_);
        packed[g] = ((unsigned)(e0 & 255)) |
                    ((unsigned)(e1 & 255) << 8) |
                    ((unsigned)(e2 & 255) << 16) |
                    ((unsigned)(e3 & 255) << 24);
    }
    v4i* dst = (v4i*)(bp + tile * 1024 + (long long)lane * 32);
    dst[0] = (v4i){(int)packed[0], (int)packed[1], (int)packed[2], (int)packed[3]};
    dst[1] = (v4i){(int)packed[4], (int)packed[5], (int)packed[6], (int)packed[7]};
}

__global__ __launch_bounds__(256) void gemm_iu8_kernel(
    const unsigned char* __restrict__ ap,
    const unsigned char* __restrict__ bp,
    const float* __restrict__ scale,
    _Float16* __restrict__ out) {

    const int tid  = threadIdx.x;
    const int lane = tid & 31;
    const int wave = tid >> 5;
    const int wm   = wave & 3;
    const int wn   = wave >> 2;
    const int b    = blockIdx.z;

    const int mt0 = blockIdx.y * 16 + wm * 4;
    const int nt0 = blockIdx.x * 8 + wn * 4;

    const unsigned char* aBase =
        ap + (((long long)b * MT16 + mt0) * KT64) * 1024 + (long long)lane * 32;
    const unsigned char* bBase =
        bp + ((long long)nt0 * KT64) * 1024 + (long long)lane * 32;

    v8i acc[4][4];
#pragma unroll
    for (int i = 0; i < 4; ++i)
#pragma unroll
        for (int j = 0; j < 4; ++j)
            acc[i][j] = (v8i){0, 0, 0, 0, 0, 0, 0, 0};

    for (int kt = 0; kt < KT64; ++kt) {
        v8i a[4];
        v8i bt[4];
#pragma unroll
        for (int i = 0; i < 4; ++i)
            a[i] = *(const v8i*)(aBase + (long long)(i * KT64 + kt) * 1024);
#pragma unroll
        for (int j = 0; j < 4; ++j)
            bt[j] = *(const v8i*)(bBase + (long long)(j * KT64 + kt) * 1024);

#pragma unroll
        for (int i = 0; i < 4; ++i) {
#pragma unroll
            for (int j = 0; j < 4; ++j) {
                acc[i][j] = __builtin_amdgcn_wmma_i32_16x16x64_iu8(
                    true, a[i], true, bt[j], acc[i][j], false, false);
            }
        }
    }

    const float s = *scale;

    const int rOff = (lane >> 4) * 8;
    const int nCol = lane & 15;
#pragma unroll
    for (int i = 0; i < 4; ++i) {
#pragma unroll
        for (int j = 0; j < 4; ++j) {
            AccU u;
            u.v = acc[i][j];
#pragma unroll
            for (int r = 0; r < 8; ++r) {
                const int m = (mt0 + i) * 16 + rOff + r;
                const int n = (nt0 + j) * 16 + nCol;
                __builtin_nontemporal_store(
                    (_Float16)((float)u.s[r] * s),
                    out + ((long long)b * M_ + m) * N_ + n);
            }
        }
    }
}

extern "C" void kernel_launch(void* const* d_in, const int* in_sizes, int n_in,
                              void* d_out, int out_size, void* d_ws, size_t ws_size,
                              hipStream_t stream) {
    (void)in_sizes; (void)n_in; (void)out_size; (void)ws_size;

    const int*   x     = (const int*)d_in[0];
    const int*   y     = (const int*)d_in[1];
    const float* scale = (const float*)d_in[2];
    _Float16*    out   = (_Float16*)d_out;

    unsigned char* ap = (unsigned char*)d_ws;
    unsigned char* bp = ap + A_PACK_BYTES;

    {
        long long threadsA = (long long)B_ * MT16 * KT64 * 32;
        pack_a_kernel<<<(int)(threadsA / 256), 256, 0, stream>>>(x, ap);
        long long threadsB = (long long)NT16 * KT64 * 32;
        pack_b_kernel<<<(int)(threadsB / 256), 256, 0, stream>>>(y, bp);
    }

    {
        dim3 grid(N_ / 128, M_ / 256, B_);
        gemm_iu8_kernel<<<grid, 256, 0, stream>>>(ap, bp, scale, out);
    }
}
